// ConceptRWKV_28278064677452
// MI455X (gfx1250) — matrix operand order repaired and checked by reading, not compiled, not run on hardware
//
#include <hip/hip_runtime.h>
#include <cstdint>
#include <cstddef>

// ---------------------------------------------------------------------------
// Model constants (fixed by the reference)
// ---------------------------------------------------------------------------
#define BB 2
#define TT 2048
#define VV 32000
#define DD 512
#define FFD 2048
#define LL 6
#define CDIM 64
#define MROWS (BB * TT)          // 4096 token rows
#define EPSV 1e-5f

#define NSEG 16                  // decay-scan segments
#define SEGLEN (TT / NSEG)       // 128

// GEMM tiling
#define BK 32                    // K-chunk staged in LDS
#define LSTR 40                  // padded LDS row stride (elements): conflict-free, 16B-mult

// ---------------------------------------------------------------------------
// CDNA5 async global->LDS copy (ASYNCcnt path), with safe fallback
// ---------------------------------------------------------------------------
#if defined(__has_builtin)
#if __has_builtin(__builtin_amdgcn_global_load_async_to_lds_b128) && \
    __has_builtin(__builtin_amdgcn_s_wait_asynccnt)
#define ASYNC_OK 1
#endif
#endif
#ifndef ASYNC_OK
#define ASYNC_OK 0
#endif

typedef __attribute__((ext_vector_type(4))) int v4i_t;
typedef __attribute__((address_space(1))) v4i_t glob_v4i_t;   // global (AS1) int4
typedef __attribute__((address_space(3))) v4i_t lds_v4i_t;    // LDS (AS3) int4

__device__ __forceinline__ void async_cp_b128(const void* g, void* l) {
#if ASYNC_OK
    // flat->AS1 is identity; flat->AS3 is truncation to the low 32 bits (LDS aperture)
    __builtin_amdgcn_global_load_async_to_lds_b128(
        (glob_v4i_t*)(uintptr_t)g,
        (lds_v4i_t*)(unsigned int)(uintptr_t)l, 0, 0);
#else
    *(uint4*)l = *(const uint4*)g;
#endif
}

__device__ __forceinline__ void wait_async_pending(bool have_next) {
#if ASYNC_OK
    if (have_next) __builtin_amdgcn_s_wait_asynccnt(4);   // chunk c done, c+1 in flight
    else           __builtin_amdgcn_s_wait_asynccnt(0);
#else
    (void)have_next;
#endif
}

// ---------------------------------------------------------------------------
// WMMA types (CDNA5 gfx1250, wave32)
// ---------------------------------------------------------------------------
typedef __attribute__((ext_vector_type(16))) __bf16 v16bf;
typedef __attribute__((ext_vector_type(8)))  float  v8f;

union FragBF {
    unsigned int u[8];
    uint4        q[2];
    v16bf        v;
};

__device__ __forceinline__ unsigned short f2bf(float f) {
    union { float f; unsigned int u; } x; x.f = f;
    unsigned int r = x.u + 0x7FFFu + ((x.u >> 16) & 1u);   // round-to-nearest-even
    return (unsigned short)(r >> 16);
}

__device__ __forceinline__ float sigmoidf_(float x) {
    return 1.0f / (1.0f + __expf(-x));
}

// ---------------------------------------------------------------------------
// LDS-staged GEMM: C[M,N] f32 = A[M,K] bf16 * Bt[N,K] bf16 (B pre-transposed)
// Block 256 thr = 8 waves, tile 128x128; wave tile 32x64 (2x4 WMMA accs).
// Async double-buffered K-chunks of 32.
// Requires M%128==0, N%128==0, K%32==0.
// EPI: 0=none, 1=sigmoid, 2=residual add (aux=[M,N]), 3=bias add (aux=[N])
// ---------------------------------------------------------------------------
template <int EPI>
__global__ void __launch_bounds__(256)
gemm_lds_k(const unsigned short* __restrict__ A,
           const unsigned short* __restrict__ Bt,
           float* __restrict__ C,
           const float* __restrict__ aux,
           int M, int N, int K) {
    __shared__ __align__(16) unsigned short sA[2][128 * LSTR];
    __shared__ __align__(16) unsigned short sB[2][128 * LSTR];

    const int lane = threadIdx.x & 31;
    const int wave = threadIdx.x >> 5;
    const int m0 = blockIdx.y * 128;
    const int n0 = blockIdx.x * 128;
    const int l15 = lane & 15;
    const int hi  = (lane >> 4) & 1;
    const int wm  = wave >> 1;      // 0..3 : 32-row strip
    const int wn  = wave & 1;       // 0..1 : 64-col strip

    // copy mapping: 32 lanes x 16B = 8 rows x 64B per instruction
    const int cprow = lane >> 2;          // 0..7
    const int cpcol = (lane & 3) * 8;     // element offset (16B granules)

    const int nc = K / BK;

    // ---- prefetch chunk 0 ----
    {
        const int k0 = 0;
#pragma unroll
        for (int j = 0; j < 2; ++j) {
            int r = wave * 16 + j * 8 + cprow;
            async_cp_b128(A  + (size_t)(m0 + r) * K + k0 + cpcol,
                          &sA[0][r * LSTR + cpcol]);
            async_cp_b128(Bt + (size_t)(n0 + r) * K + k0 + cpcol,
                          &sB[0][r * LSTR + cpcol]);
        }
    }

    v8f acc[2][4] = {};

    for (int c = 0; c < nc; ++c) {
        const int buf = c & 1;
        const bool have_next = (c + 1) < nc;
        if (have_next) {
            const int k0 = (c + 1) * BK;
            const int nb = buf ^ 1;
#pragma unroll
            for (int j = 0; j < 2; ++j) {
                int r = wave * 16 + j * 8 + cprow;
                async_cp_b128(A  + (size_t)(m0 + r) * K + k0 + cpcol,
                              &sA[nb][r * LSTR + cpcol]);
                async_cp_b128(Bt + (size_t)(n0 + r) * K + k0 + cpcol,
                              &sB[nb][r * LSTR + cpcol]);
            }
        }
        wait_async_pending(have_next);
        __syncthreads();

        // ---- compute one K-step of 32 from LDS ----
        FragBF a[2];
#pragma unroll
        for (int ms = 0; ms < 2; ++ms) {
            const unsigned short* pa =
                &sA[buf][(wm * 32 + ms * 16 + l15) * LSTR + hi * 8];
            a[ms].q[0] = *(const uint4*)pa;          // K = khalf + 0..7
            a[ms].q[1] = *(const uint4*)(pa + 16);   // K = khalf + 16..23
        }
#pragma unroll
        for (int ns = 0; ns < 4; ++ns) {
            FragBF b;
            const unsigned short* pb =
                &sB[buf][(wn * 64 + ns * 16 + l15) * LSTR + hi * 8];
            b.q[0] = *(const uint4*)pb;              // K = khalf + 0..7
            b.q[1] = *(const uint4*)(pb + 16);       // K = khalf + 16..23
            acc[0][ns] = __builtin_amdgcn_wmma_f32_16x16x32_bf16(
                false, a[0].v, false, b.v, (short)0, acc[0][ns], false, false);
            acc[1][ns] = __builtin_amdgcn_wmma_f32_16x16x32_bf16(
                false, a[1].v, false, b.v, (short)0, acc[1][ns], false, false);
        }
        __syncthreads();
    }

    // ---- epilogue ----
#pragma unroll
    for (int ms = 0; ms < 2; ++ms) {
#pragma unroll
        for (int ns = 0; ns < 4; ++ns) {
#pragma unroll
            for (int j = 0; j < 8; ++j) {
                int m = m0 + wm * 32 + ms * 16 + hi * 8 + j;
                int n = n0 + wn * 64 + ns * 16 + l15;
                float v = acc[ms][ns][j];
                if (EPI == 1) v = sigmoidf_(v);
                if (EPI == 2) v += aux[(size_t)m * N + n];
                if (EPI == 3) v += aux[n];
                C[(size_t)m * N + n] = v;
            }
        }
    }
}

// ---------------------------------------------------------------------------
// Direct-global GEMM (for the tiny encoder matmul, N=64)
// Wave computes 16x64; block = 4 waves -> 64x64 tile. Requires N%64, K%32.
// ---------------------------------------------------------------------------
template <int EPI>
__global__ void __launch_bounds__(128)
gemm_direct_k(const unsigned short* __restrict__ A,
              const unsigned short* __restrict__ Bt,
              float* __restrict__ C,
              const float* __restrict__ aux,
              int M, int N, int K) {
    const int lane = threadIdx.x & 31;
    const int wave = threadIdx.x >> 5;
    const int m0 = (blockIdx.y * 4 + wave) * 16;
    const int n0 = blockIdx.x * 64;
    if (m0 >= M) return;

    const int l15    = lane & 15;
    const int hi     = (lane >> 4) & 1;
    const int khalfA = hi * 8;
    const int khalfB = hi * 8;

    const unsigned int* Arow =
        (const unsigned int*)(A + (size_t)(m0 + l15) * (size_t)K);

    v8f acc[4] = {};
    for (int k0 = 0; k0 < K; k0 += 32) {
        FragBF af;
#pragma unroll
        for (int i = 0; i < 8; ++i) {
            int k = k0 + khalfA + ((i < 4) ? (2 * i) : (8 + 2 * i));
            af.u[i] = Arow[k >> 1];
        }
#pragma unroll
        for (int t = 0; t < 4; ++t) {
            const unsigned int* Brow =
                (const unsigned int*)(Bt + (size_t)(n0 + 16 * t + l15) * (size_t)K);
            FragBF bfr;
#pragma unroll
            for (int i = 0; i < 8; ++i) {
                int k = k0 + khalfB + ((i < 4) ? (2 * i) : (8 + 2 * i));
                bfr.u[i] = Brow[k >> 1];
            }
            acc[t] = __builtin_amdgcn_wmma_f32_16x16x32_bf16(
                false, af.v, false, bfr.v, (short)0, acc[t], false, false);
        }
    }
    const int mo = hi * 8;
#pragma unroll
    for (int t = 0; t < 4; ++t) {
#pragma unroll
        for (int j = 0; j < 8; ++j) {
            int m = m0 + mo + j;
            int n = n0 + 16 * t + l15;
            float v = acc[t][j];
            if (EPI == 1) v = sigmoidf_(v);
            if (EPI == 2) v += aux[(size_t)m * N + n];
            if (EPI == 3) v += aux[n];
            C[(size_t)m * N + n] = v;
        }
    }
}

// ---------------------------------------------------------------------------
// Transpose + f32 -> bf16 convert:  in[R,Ccols] f32  ->  out[Ccols,R] bf16
// ---------------------------------------------------------------------------
__global__ void transpose_cvt_k(const float* __restrict__ in,
                                unsigned short* __restrict__ out,
                                int R, int Ccols) {
    __shared__ float tile[32][33];
    int rb = blockIdx.y * 32, cb = blockIdx.x * 32;
    for (int i = threadIdx.y; i < 32; i += 8) {
        int r = rb + i, c = cb + threadIdx.x;
        tile[i][threadIdx.x] = (r < R && c < Ccols) ? in[(size_t)r * Ccols + c] : 0.f;
    }
    __syncthreads();
    for (int i = threadIdx.y; i < 32; i += 8) {
        int c = cb + i, r = rb + threadIdx.x;
        if (c < Ccols && r < R)
            out[(size_t)c * R + r] = f2bf(tile[threadIdx.x][i]);
    }
}

__global__ void cvt_bf16_k(const float* __restrict__ in,
                           unsigned short* __restrict__ out, size_t n) {
    for (size_t i = (size_t)blockIdx.x * blockDim.x + threadIdx.x; i < n;
         i += (size_t)gridDim.x * blockDim.x)
        out[i] = f2bf(in[i]);
}

__global__ void silu_gate_bf16_k(const float* __restrict__ a,
                                 const float* __restrict__ c,
                                 unsigned short* __restrict__ out, size_t n) {
    for (size_t i = (size_t)blockIdx.x * blockDim.x + threadIdx.x; i < n;
         i += (size_t)gridDim.x * blockDim.x) {
        float av = a[i];
        out[i] = f2bf(av * sigmoidf_(av) * c[i]);
    }
}

// ---------------------------------------------------------------------------
// LayerNorm over D=512, one wave32 per row (gathers embedding if idx!=nullptr)
// ---------------------------------------------------------------------------
__global__ void ln_rows_k(const float* __restrict__ in,
                          const int* __restrict__ idx,
                          const float* __restrict__ emb,
                          const float* __restrict__ g,
                          const float* __restrict__ b,
                          float* __restrict__ out, int rows) {
    int wave = threadIdx.x >> 5, lane = threadIdx.x & 31;
    int row = blockIdx.x * (blockDim.x >> 5) + wave;
    if (row >= rows) return;
    const float* p = idx ? (emb + (size_t)idx[row] * DD)
                         : (in + (size_t)row * DD);
    float v[16];
    float s = 0.f;
#pragma unroll
    for (int i = 0; i < 16; ++i) { v[i] = p[lane + 32 * i]; s += v[i]; }
#pragma unroll
    for (int off = 16; off; off >>= 1) s += __shfl_xor(s, off, 32);
    float mean = s * (1.0f / DD);
    float vs = 0.f;
#pragma unroll
    for (int i = 0; i < 16; ++i) { float d = v[i] - mean; vs += d * d; }
#pragma unroll
    for (int off = 16; off; off >>= 1) vs += __shfl_xor(vs, off, 32);
    float inv = rsqrtf(vs * (1.0f / DD) + EPSV);
    float* o = out + (size_t)row * DD;
#pragma unroll
    for (int i = 0; i < 16; ++i) {
        int c = lane + 32 * i;
        o[c] = (v[i] - mean) * inv * g[c] + b[c];
    }
}

// ---------------------------------------------------------------------------
// Segmented decay scan:  s_t = w*s_{t-1} + k_t*v_t ; out_t = r_t*s_t
// pass1: per-(b,d,seg) partial end-state from zero start
// pass2: per-(b,d) scan of carries across NSEG segments
// pass3: per-(b,d,seg) rescan with carry-in, write out
// ---------------------------------------------------------------------------
__device__ __forceinline__ float decay_w(const float* dec, int d) {
    return fmaxf(sigmoidf_(dec[d]), 1e-7f);
}

__global__ void scan_pass1_k(const float* __restrict__ kb,
                             const float* __restrict__ vb,
                             const float* __restrict__ dec,
                             float* __restrict__ segend) {
    int i = blockIdx.x * blockDim.x + threadIdx.x;            // (b, seg, d)
    if (i >= BB * NSEG * DD) return;
    int d = i % DD;
    int seg = (i / DD) % NSEG;
    int b = i / (DD * NSEG);
    float w = decay_w(dec, d);
    float s = 0.f;
    size_t base = (size_t)b * TT * DD + (size_t)seg * SEGLEN * DD + d;
    for (int t = 0; t < SEGLEN; ++t) {
        size_t o = base + (size_t)t * DD;
        s = s * w + kb[o] * vb[o];
    }
    segend[i] = s;
}

__global__ void scan_pass2_k(const float* __restrict__ segend,
                             const float* __restrict__ dec,
                             float* __restrict__ carryin) {
    int i = blockIdx.x * blockDim.x + threadIdx.x;            // (b, d)
    if (i >= BB * DD) return;
    int d = i % DD, b = i / DD;
    float w = decay_w(dec, d);
    float wL = w;
#pragma unroll
    for (int j = 0; j < 7; ++j) wL *= wL;                     // w^128
    float s = 0.f;
    for (int seg = 0; seg < NSEG; ++seg) {
        size_t o = (size_t)b * NSEG * DD + (size_t)seg * DD + d;
        carryin[o] = s;                                       // state entering segment
        s = segend[o] + wL * s;
    }
}

__global__ void scan_pass3_k(const float* __restrict__ kb,
                             const float* __restrict__ vb,
                             const float* __restrict__ rb,
                             const float* __restrict__ dec,
                             const float* __restrict__ carryin,
                             float* __restrict__ out) {
    int i = blockIdx.x * blockDim.x + threadIdx.x;            // (b, seg, d)
    if (i >= BB * NSEG * DD) return;
    int d = i % DD;
    int seg = (i / DD) % NSEG;
    int b = i / (DD * NSEG);
    float w = decay_w(dec, d);
    float s = carryin[(size_t)b * NSEG * DD + (size_t)seg * DD + d];
    size_t base = (size_t)b * TT * DD + (size_t)seg * SEGLEN * DD + d;
    for (int t = 0; t < SEGLEN; ++t) {
        size_t o = base + (size_t)t * DD;
        s = s * w + kb[o] * vb[o];
        out[o] = rb[o] * s;
    }
}

// ---------------------------------------------------------------------------
// GroupNorm(1, D) over (T,D) per sample: two-pass with atomics.
// ---------------------------------------------------------------------------
__global__ void zero_k(float* p, int n) {
    int i = blockIdx.x * blockDim.x + threadIdx.x;
    if (i < n) p[i] = 0.f;
}

__device__ __forceinline__ float block_sum(float v, float* sm) {
    __syncthreads();
#pragma unroll
    for (int off = 16; off; off >>= 1) v += __shfl_xor(v, off, 32);
    int lane = threadIdx.x & 31, w = threadIdx.x >> 5;
    if (lane == 0) sm[w] = v;
    __syncthreads();
    int nw = blockDim.x >> 5;
    v = (threadIdx.x < nw) ? sm[threadIdx.x] : 0.f;
    if (w == 0) {
#pragma unroll
        for (int off = 16; off; off >>= 1) v += __shfl_xor(v, off, 32);
    }
    return v;   // valid in thread 0
}

__global__ void gn_reduce_k(const float* __restrict__ x, float* __restrict__ red) {
    __shared__ float sm[8];
    const size_t per_b = (size_t)TT * DD;
    size_t base = (size_t)blockIdx.y * per_b;
    float s = 0.f, s2 = 0.f;
    for (size_t i = (size_t)blockIdx.x * blockDim.x + threadIdx.x; i < per_b;
         i += (size_t)gridDim.x * blockDim.x) {
        float v = x[base + i];
        s += v; s2 += v * v;
    }
    float ts = block_sum(s, sm);
    float ts2 = block_sum(s2, sm);
    if (threadIdx.x == 0) {
        atomicAdd(&red[2 * blockIdx.y + 0], ts);
        atomicAdd(&red[2 * blockIdx.y + 1], ts2);
    }
}

__global__ void gn_apply_k(float* __restrict__ x, const float* __restrict__ red,
                           const float* __restrict__ g, const float* __restrict__ b) {
    const size_t per_b = (size_t)TT * DD;
    const float invn = 1.0f / (float)per_b;
    int bi = blockIdx.y;
    float m = red[2 * bi] * invn;
    float var = red[2 * bi + 1] * invn - m * m;
    float inv = rsqrtf(var + EPSV);
    size_t base = (size_t)bi * per_b;
    for (size_t i = (size_t)blockIdx.x * blockDim.x + threadIdx.x; i < per_b;
         i += (size_t)gridDim.x * blockDim.x) {
        int d = (int)(i % DD);
        x[base + i] = (x[base + i] - m) * inv * g[d] + b[d];
    }
}

// ---------------------------------------------------------------------------
// Host orchestration
// ---------------------------------------------------------------------------
extern "C" void kernel_launch(void* const* d_in, const int* in_sizes, int n_in,
                              void* d_out, int out_size, void* d_ws, size_t ws_size,
                              hipStream_t stream) {
    (void)in_sizes; (void)n_in; (void)out_size; (void)ws_size;

    const int*   idx     = (const int*)  d_in[0];
    const float* embed   = (const float*)d_in[1];
    const float* ln_in_g = (const float*)d_in[2];
    const float* ln_in_b = (const float*)d_in[3];
    const float* ln1_g   = (const float*)d_in[4];
    const float* ln1_b   = (const float*)d_in[5];
    const float* Wr      = (const float*)d_in[6];
    const float* Wk      = (const float*)d_in[7];
    const float* Wv      = (const float*)d_in[8];
    const float* Wo_tm   = (const float*)d_in[9];
    const float* decay   = (const float*)d_in[10];
    const float* gn_g    = (const float*)d_in[11];
    const float* gn_b    = (const float*)d_in[12];
    const float* ln2_g   = (const float*)d_in[13];
    const float* ln2_b   = (const float*)d_in[14];
    const float* W1      = (const float*)d_in[15];
    const float* W2      = (const float*)d_in[16];
    const float* Wo_ff   = (const float*)d_in[17];
    const float* ln_out_g= (const float*)d_in[18];
    const float* ln_out_b= (const float*)d_in[19];
    const float* enc_W   = (const float*)d_in[20];
    const float* enc_b   = (const float*)d_in[21];
    const float* dec_W   = (const float*)d_in[22];
    const float* dec_b   = (const float*)d_in[23];
    float* logits = (float*)d_out;

    size_t off = 0;
    auto alloc = [&](size_t bytes) -> void* {
        off = (off + 255) & ~(size_t)255;
        void* p = (char*)d_ws + off;
        off += bytes;
        return p;
    };
    const size_t MD = (size_t)MROWS * DD;
    const size_t MF = (size_t)MROWS * FFD;

    float* hA    = (float*)alloc(MD * 4);
    float* hBuf  = (float*)alloc(MD * 4);
    float* xbuf  = (float*)alloc(MD * 4);
    float* rbuf  = (float*)alloc(MD * 4);
    float* kbuf  = (float*)alloc(MD * 4);
    float* vbuf  = (float*)alloc(MD * 4);
    float* att   = (float*)alloc(MD * 4);
    float* ff1   = (float*)alloc(MF * 4);
    float* ff2   = (float*)alloc(MF * 4);
    float* conc  = (float*)alloc((size_t)MROWS * CDIM * 4);
    float* gnred = (float*)alloc(4 * sizeof(float));
    float* segend  = (float*)alloc((size_t)BB * NSEG * DD * 4);
    float* carryin = (float*)alloc((size_t)BB * NSEG * DD * 4);

    unsigned short* actbf = (unsigned short*)alloc(MF * 2);
    unsigned short* WrT   = (unsigned short*)alloc((size_t)LL * DD * DD * 2);
    unsigned short* WkT   = (unsigned short*)alloc((size_t)LL * DD * DD * 2);
    unsigned short* WvT   = (unsigned short*)alloc((size_t)LL * DD * DD * 2);
    unsigned short* WoT   = (unsigned short*)alloc((size_t)LL * DD * DD * 2);
    unsigned short* W1T   = (unsigned short*)alloc((size_t)LL * DD * FFD * 2);
    unsigned short* W2T   = (unsigned short*)alloc((size_t)LL * DD * FFD * 2);
    unsigned short* WofT  = (unsigned short*)alloc((size_t)LL * FFD * DD * 2);
    unsigned short* encT  = (unsigned short*)alloc((size_t)DD * CDIM * 2);
    unsigned short* decT  = (unsigned short*)alloc((size_t)CDIM * VV * 2);

    auto transpose = [&](const float* in, unsigned short* out, int R, int C) {
        dim3 grid((C + 31) / 32, (R + 31) / 32), block(32, 8);
        transpose_cvt_k<<<grid, block, 0, stream>>>(in, out, R, C);
    };
    auto gemm = [&](const unsigned short* A, const unsigned short* Bt, float* C,
                    int M, int N, int K, int epi, const float* aux) {
        if ((N % 128) == 0) {
            dim3 grid(N / 128, M / 128), block(256);
            switch (epi) {
            case 0: gemm_lds_k<0><<<grid, block, 0, stream>>>(A, Bt, C, aux, M, N, K); break;
            case 1: gemm_lds_k<1><<<grid, block, 0, stream>>>(A, Bt, C, aux, M, N, K); break;
            case 2: gemm_lds_k<2><<<grid, block, 0, stream>>>(A, Bt, C, aux, M, N, K); break;
            default:gemm_lds_k<3><<<grid, block, 0, stream>>>(A, Bt, C, aux, M, N, K); break;
            }
        } else {
            dim3 grid(N / 64, M / 64), block(128);
            switch (epi) {
            case 0: gemm_direct_k<0><<<grid, block, 0, stream>>>(A, Bt, C, aux, M, N, K); break;
            case 1: gemm_direct_k<1><<<grid, block, 0, stream>>>(A, Bt, C, aux, M, N, K); break;
            case 2: gemm_direct_k<2><<<grid, block, 0, stream>>>(A, Bt, C, aux, M, N, K); break;
            default:gemm_direct_k<3><<<grid, block, 0, stream>>>(A, Bt, C, aux, M, N, K); break;
            }
        }
    };
    auto cvt = [&](const float* in, unsigned short* out, size_t n) {
        int blocks = (int)((n + 255) / 256); if (blocks > 8192) blocks = 8192;
        cvt_bf16_k<<<blocks, 256, 0, stream>>>(in, out, n);
    };
    auto ln = [&](const float* in, const float* g, const float* b, float* out) {
        ln_rows_k<<<MROWS / 8, 256, 0, stream>>>(in, nullptr, nullptr, g, b, out, MROWS);
    };

    // ---- 1. weights -> bf16 transposed ([N,K]); full bf16 set fits in 192MB L2 ----
    for (int l = 0; l < LL; ++l) {
        size_t wdd = (size_t)l * DD * DD, wdf = (size_t)l * DD * FFD;
        transpose(Wr    + wdd, WrT  + wdd, DD, DD);
        transpose(Wk    + wdd, WkT  + wdd, DD, DD);
        transpose(Wv    + wdd, WvT  + wdd, DD, DD);
        transpose(Wo_tm + wdd, WoT  + wdd, DD, DD);
        transpose(W1    + wdf, W1T  + wdf, DD, FFD);
        transpose(W2    + wdf, W2T  + wdf, DD, FFD);
        transpose(Wo_ff + wdf, WofT + wdf, FFD, DD);
    }
    transpose(enc_W, encT, DD, CDIM);
    transpose(dec_W, decT, CDIM, VV);

    // ---- 2. embedding gather + input LN ----
    ln_rows_k<<<MROWS / 8, 256, 0, stream>>>(nullptr, idx, embed,
                                             ln_in_g, ln_in_b, hA, MROWS);

    float* hcur = hA;
    float* hnxt = hBuf;

    // ---- 3. layers ----
    for (int l = 0; l < LL; ++l) {
        size_t wdd = (size_t)l * DD * DD, wdf = (size_t)l * DD * FFD;
        size_t vd  = (size_t)l * DD;

        // TimeMix
        ln(hcur, ln1_g + vd, ln1_b + vd, xbuf);
        cvt(xbuf, actbf, MD);
        gemm(actbf, WrT + wdd, rbuf, MROWS, DD, DD, 1, nullptr);   // r = sigmoid(x Wr)
        gemm(actbf, WkT + wdd, kbuf, MROWS, DD, DD, 0, nullptr);
        gemm(actbf, WvT + wdd, vbuf, MROWS, DD, DD, 0, nullptr);

        scan_pass1_k<<<(BB * NSEG * DD + 255) / 256, 256, 0, stream>>>(
            kbuf, vbuf, decay + vd, segend);
        scan_pass2_k<<<(BB * DD + 255) / 256, 256, 0, stream>>>(
            segend, decay + vd, carryin);
        scan_pass3_k<<<(BB * NSEG * DD + 255) / 256, 256, 0, stream>>>(
            kbuf, vbuf, rbuf, decay + vd, carryin, att);

        // GroupNorm over (T,D) per sample
        zero_k<<<1, 32, 0, stream>>>(gnred, 2 * BB);
        { dim3 g(256, BB); gn_reduce_k<<<g, 256, 0, stream>>>(att, gnred); }
        { dim3 g(256, BB); gn_apply_k<<<g, 256, 0, stream>>>(att, gnred,
                                                             gn_g + vd, gn_b + vd); }
        cvt(att, actbf, MD);
        gemm(actbf, WoT + wdd, hnxt, MROWS, DD, DD, 2, hcur);      // h += out Wo_tm
        { float* t = hcur; hcur = hnxt; hnxt = t; }

        // ChannelMix
        ln(hcur, ln2_g + vd, ln2_b + vd, xbuf);
        cvt(xbuf, actbf, MD);
        gemm(actbf, W1T + wdf, ff1, MROWS, FFD, DD, 0, nullptr);
        gemm(actbf, W2T + wdf, ff2, MROWS, FFD, DD, 0, nullptr);
        silu_gate_bf16_k<<<8192, 256, 0, stream>>>(ff1, ff2, actbf, MF);
        gemm(actbf, WofT + wdf, hnxt, MROWS, DD, FFD, 2, hcur);    // h += gate Wo_ff
        { float* t = hcur; hcur = hnxt; hnxt = t; }
    }

    // ---- 4. head ----
    ln(hcur, ln_out_g, ln_out_b, xbuf);
    cvt(xbuf, actbf, MD);
    gemm(actbf, encT, conc, MROWS, CDIM, DD, 3, enc_b);            // concepts (N=64 -> direct)
    cvt(conc, actbf, (size_t)MROWS * CDIM);
    gemm(actbf, decT, logits, MROWS, VV, CDIM, 3, dec_b);          // logits
}
